// MultiHeadAttention_29248727286084
// MI455X (gfx1250) — matrix operand order repaired and checked by reading, not compiled, not run on hardware
//
#include <hip/hip_runtime.h>

// ---------- CDNA5 WMMA types ----------
typedef __attribute__((ext_vector_type(16))) _Float16 v16h;
typedef __attribute__((ext_vector_type(8)))  float    v8f;

#define LWIN  49
#define LP    64
#define CDIM  128
#define HNUM  4
#define DHEAD 32
#define NC3   384
#define WNUM  64
#define LL    (LWIN * LWIN)   // 2401

union AB16 { v16h v; unsigned int u[8]; };

// A operand: 16x32 f16, element (m,k) at base[m*rowStride + k] (row-major over k).
// Lanes 0-15/16-31 both M=lane%16; v0..7 -> K = 16*hi + 2v (contiguous, matches lds_load_BT).
__device__ __forceinline__ v16h lds_load_A(const _Float16* base, int rowStride,
                                           int m0, int k0, int lane) {
    AB16 r;
    const int m  = m0 + (lane & 15);
    const int hi = (lane >> 4) & 1;
    const _Float16* p = base + m * rowStride + k0;
#pragma unroll
    for (int v = 0; v < 4; ++v)
        r.u[v]     = *(const unsigned int*)(p + 16 * hi + 2 * v);
#pragma unroll
    for (int v = 0; v < 4; ++v)
        r.u[4 + v] = *(const unsigned int*)(p + 16 * hi + 8 + 2 * v);
    return r.v;
}

// B operand: 32x16 f16, element (k,n) stored TRANSPOSED at base[n*rowStride + k].
// Lanes 0-15 K=0..15 (v holds K=2v,2v+1); lanes 16-31 K=16..31.
__device__ __forceinline__ v16h lds_load_BT(const _Float16* base, int rowStride,
                                            int n0, int k0, int lane) {
    AB16 r;
    const int n  = n0 + (lane & 15);
    const int kb = k0 + 16 * ((lane >> 4) & 1);
    const _Float16* p = base + n * rowStride + kb;
#pragma unroll
    for (int v = 0; v < 8; ++v)
        r.u[v] = *(const unsigned int*)(p + 2 * v);
    return r.v;
}

// Vectorized f32 -> f16 staging: float4 global loads, packed b64 LDS stores.
__device__ __forceinline__ void stage_w(const float* __restrict__ src, _Float16* dst,
                                        int n, int tid) {
    for (int i = tid * 4; i < n; i += 256 * 4) {
        const float4 f = *(const float4*)(src + i);
        union { _Float16 h[4]; unsigned long long u; } p;
        p.h[0] = (_Float16)f.x; p.h[1] = (_Float16)f.y;
        p.h[2] = (_Float16)f.z; p.h[3] = (_Float16)f.w;
        *(unsigned long long*)(dst + i) = p.u;
    }
}

// LDS map (bytes), one fused window per workgroup (~231 KB of 320 KB WGP LDS):
//   [0      ,  16384)  xA   [64][128] f16  (A for QKV)  -- reused as O-stage
//   [16384  , 114688)  wQKV [384][128] f16              -- reused as P [4][64][64] f16
//   [114688 , 131072)  qA   [4][64][32] f16 (A for S)
//   [131072 , 147456)  kB   [4][64][32] f16 (B^T for S)
//   [147456 , 163840)  vT   [4][32][64] f16 (B^T for PV)
//   [163840 , 196608)  wP   [128][128] f16 (B^T for proj)
//   [196608 , 198144)  bQ   [384] f32
//   [198144 , 198656)  bP   [128] f32
//   [198656 , 237072)  bmS  [4][49*49] f32  (bias[h]+mask[wIdx], combined)
#define SMEM_FLOATS (237072 / 4)

__global__ __launch_bounds__(256, 1)
void swin_mha_fused(const float* __restrict__ x,      const float* __restrict__ mask,
                    const float* __restrict__ qkv_w,  const float* __restrict__ qkv_b,
                    const float* __restrict__ bias,   const float* __restrict__ proj_w,
                    const float* __restrict__ proj_b, float* __restrict__ out) {
    __shared__ float smemf[SMEM_FLOATS];
    char* smem = (char*)smemf;
    _Float16* xA   = (_Float16*)(smem);
    _Float16* wQKV = (_Float16*)(smem + 16384);
    _Float16* Pm   = wQKV;                         // alias after QKV phase
    _Float16* qA   = (_Float16*)(smem + 114688);
    _Float16* kB   = (_Float16*)(smem + 131072);
    _Float16* vT   = (_Float16*)(smem + 147456);
    _Float16* wP   = (_Float16*)(smem + 163840);
    float*    bQ   = (float*)(smem + 196608);
    float*    bP   = (float*)(smem + 198144);
    float*    bmS  = (float*)(smem + 198656);
    _Float16* Ost  = xA;                           // alias after QKV phase

    const int tid  = threadIdx.x;
    const int lane = tid & 31;
    // force wave id into an SGPR so region/tile selection is scalar (s_cbranch, no EXEC juggling)
    const int wave = __builtin_amdgcn_readfirstlane(tid >> 5);
    const int b    = blockIdx.x;
    const int wIdx = b & (WNUM - 1);
    const float scale = 0.1767766952966369f;       // 32^-0.5

    // ---- Stage inputs to LDS (vectorized), zero-pad x rows 49..63, combine bias+mask ----
    const float* xb = x + (size_t)b * LWIN * CDIM;
    stage_w(xb, xA, LWIN * CDIM, tid);
    for (int i = LWIN * CDIM + tid * 4; i < LP * CDIM; i += 1024)
        *(unsigned long long*)(xA + i) = 0ull;
    stage_w(qkv_w,  wQKV, NC3 * CDIM,  tid);
    stage_w(proj_w, wP,   CDIM * CDIM, tid);
    {
        const float* maskW = mask + (size_t)wIdx * LL;
        for (int i = tid; i < LL; i += 256) {
            const float mv = maskW[i];
#pragma unroll
            for (int h = 0; h < HNUM; ++h)
                bmS[h * LL + i] = bias[h * LL + i] + mv;
        }
    }
    for (int i = tid; i < NC3; i += 256) bQ[i] = qkv_b[i];
    if (tid < CDIM)                      bP[tid] = proj_b[tid];
    __syncthreads();

    // ---- Phase 1: QKV = x @ qkv_w^T + qkv_b -> q (scaled) / k / v^T staging ----
    // Each wave owns 3 consecutive N-tiles (q/k/v region wave-uniform) x all 4 M-tiles.
#pragma unroll 1
    for (int ntl = 0; ntl < 3; ++ntl) {
        const int nt = wave * 3 + ntl;             // 0..23 (scalar)
        v16h bf0 = lds_load_BT(wQKV, CDIM, nt * 16, 0,  lane);
        v16h bf1 = lds_load_BT(wQKV, CDIM, nt * 16, 32, lane);
        v16h bf2 = lds_load_BT(wQKV, CDIM, nt * 16, 64, lane);
        v16h bf3 = lds_load_BT(wQKV, CDIM, nt * 16, 96, lane);
        const int n   = nt * 16 + (lane & 15);
        const int hi  = (lane >> 4) & 1;
        const float bn = bQ[n];
        const int region = nt >> 3;                // 0=q, 1=k, 2=v (scalar)
        const int f = n & 127, h = f >> 5, d = f & 31;
#pragma unroll 1
        for (int mt = 0; mt < 4; ++mt) {
            // batch all A-fragment loads, then the 4-WMMA chain (partial dscnt waits)
            const v16h a0 = lds_load_A(xA, CDIM, mt * 16, 0,  lane);
            const v16h a1 = lds_load_A(xA, CDIM, mt * 16, 32, lane);
            const v16h a2 = lds_load_A(xA, CDIM, mt * 16, 64, lane);
            const v16h a3 = lds_load_A(xA, CDIM, mt * 16, 96, lane);
            v8f acc = {};
            acc = __builtin_amdgcn_wmma_f32_16x16x32_f16(false, a0, false, bf0, (short)0, acc, false, false);
            acc = __builtin_amdgcn_wmma_f32_16x16x32_f16(false, a1, false, bf1, (short)0, acc, false, false);
            acc = __builtin_amdgcn_wmma_f32_16x16x32_f16(false, a2, false, bf2, (short)0, acc, false, false);
            acc = __builtin_amdgcn_wmma_f32_16x16x32_f16(false, a3, false, bf3, (short)0, acc, false, false);
            const int mBase = mt * 16 + 8 * hi;
            if (region == 0) {                     // q, pre-scaled
                _Float16* qp = &qA[(h * LP + mBase) * DHEAD + d];
#pragma unroll
                for (int j = 0; j < 8; ++j)
                    qp[j * DHEAD] = (_Float16)((acc[j] + bn) * scale);
            } else if (region == 1) {              // k
                _Float16* kp = &kB[(h * LP + mBase) * DHEAD + d];
#pragma unroll
                for (int j = 0; j < 8; ++j)
                    kp[j * DHEAD] = (_Float16)(acc[j] + bn);
            } else {                               // v, transposed; contiguous -> b128 store
                _Float16* vp = &vT[(h * DHEAD + d) * LP + mBase];
#pragma unroll
                for (int j = 0; j < 8; j += 2) {
                    union { _Float16 h2[2]; unsigned int u; } p;
                    p.h2[0] = (_Float16)(acc[j]     + bn);
                    p.h2[1] = (_Float16)(acc[j + 1] + bn);
                    *(unsigned int*)(vp + j) = p.u;
                }
            }
        }
    }
    __syncthreads();

    // ---- Phase 2a: S = q k^T + bias + mask, row softmax, P -> LDS ----
#pragma unroll 1
    for (int s = 0; s < 2; ++s) {
        const int strip = wave * 2 + s;            // 16 strips = 4 heads x 4 Mt (scalar)
        const int h  = strip >> 2;
        const int mt = strip & 3;
        const v16h aq  = lds_load_A (qA + h * LP * DHEAD, DHEAD, mt * 16, 0, lane);
        const v16h bk0 = lds_load_BT(kB + h * LP * DHEAD, DHEAD, 0,  0, lane);
        const v16h bk1 = lds_load_BT(kB + h * LP * DHEAD, DHEAD, 16, 0, lane);
        const v16h bk2 = lds_load_BT(kB + h * LP * DHEAD, DHEAD, 32, 0, lane);
        const v16h bk3 = lds_load_BT(kB + h * LP * DHEAD, DHEAD, 48, 0, lane);
        v8f Sv[4];
        {
            v8f z = {};
            Sv[0] = __builtin_amdgcn_wmma_f32_16x16x32_f16(false, aq, false, bk0, (short)0, z, false, false);
            Sv[1] = __builtin_amdgcn_wmma_f32_16x16x32_f16(false, aq, false, bk1, (short)0, z, false, false);
            Sv[2] = __builtin_amdgcn_wmma_f32_16x16x32_f16(false, aq, false, bk2, (short)0, z, false, false);
            Sv[3] = __builtin_amdgcn_wmma_f32_16x16x32_f16(false, aq, false, bk3, (short)0, z, false, false);
        }
        const int hi = (lane >> 4) & 1;
        const int cB = lane & 15;
        const float* bmH = bmS + h * LL;
        // branch-free bias+mask add from LDS table: clamped-index ds loads + selects
#pragma unroll
        for (int nt = 0; nt < 4; ++nt) {
            const int c  = nt * 16 + cB;
            const int cc = (c < LWIN) ? c : (LWIN - 1);
#pragma unroll
            for (int j = 0; j < 8; ++j) {
                const int r  = mt * 16 + j + 8 * hi;
                const int rr = (r < LWIN) ? r : (LWIN - 1);
                const float add = bmH[rr * LWIN + cc];
                float sv = Sv[nt][j];
                sv = (r < LWIN) ? sv + add : sv;
                sv = (c < LWIN) ? sv : -1e30f;     // mask padded keys -> exp() == 0
                Sv[nt][j] = sv;
            }
        }
        // softmax per row: 64 cols = 4 tiles (per-lane) x 16 lanes (half-wave groups)
#pragma unroll
        for (int j = 0; j < 8; ++j) {
            float mx = fmaxf(fmaxf(Sv[0][j], Sv[1][j]), fmaxf(Sv[2][j], Sv[3][j]));
#pragma unroll
            for (int off = 8; off >= 1; off >>= 1)
                mx = fmaxf(mx, __shfl_xor(mx, off, 32));
            const float e0 = __expf(Sv[0][j] - mx), e1 = __expf(Sv[1][j] - mx);
            const float e2 = __expf(Sv[2][j] - mx), e3 = __expf(Sv[3][j] - mx);
            float sum = e0 + e1 + e2 + e3;
#pragma unroll
            for (int off = 8; off >= 1; off >>= 1)
                sum += __shfl_xor(sum, off, 32);
            const float inv = 1.f / sum;
            Sv[0][j] = e0 * inv; Sv[1][j] = e1 * inv;
            Sv[2][j] = e2 * inv; Sv[3][j] = e3 * inv;
        }
#pragma unroll
        for (int nt = 0; nt < 4; ++nt) {
            const int c = nt * 16 + cB;
            _Float16* pp = &Pm[(h * LP + mt * 16 + 8 * hi) * LP + c];
#pragma unroll
            for (int j = 0; j < 8; ++j)
                pp[j * LP] = (_Float16)Sv[nt][j];
        }
    }
    __syncthreads();

    // ---- Phase 2b: O = P @ v -> O-stage [64][128] f16 ----
#pragma unroll 1
    for (int s = 0; s < 2; ++s) {
        const int strip = wave * 2 + s;
        const int h  = strip >> 2;
        const int mt = strip & 3;
        const v16h a0  = lds_load_A (Pm + h * LP * LP,    LP, mt * 16, 0,  lane);
        const v16h a1  = lds_load_A (Pm + h * LP * LP,    LP, mt * 16, 32, lane);
        const v16h bv0 = lds_load_BT(vT + h * DHEAD * LP, LP, 0,  0,  lane);
        const v16h bv1 = lds_load_BT(vT + h * DHEAD * LP, LP, 0,  32, lane);
        const v16h bv2 = lds_load_BT(vT + h * DHEAD * LP, LP, 16, 0,  lane);
        const v16h bv3 = lds_load_BT(vT + h * DHEAD * LP, LP, 16, 32, lane);
        const int hi = (lane >> 4) & 1;
        v8f acc0 = {}, acc1 = {};
        acc0 = __builtin_amdgcn_wmma_f32_16x16x32_f16(false, a0, false, bv0, (short)0, acc0, false, false);
        acc0 = __builtin_amdgcn_wmma_f32_16x16x32_f16(false, a1, false, bv1, (short)0, acc0, false, false);
        acc1 = __builtin_amdgcn_wmma_f32_16x16x32_f16(false, a0, false, bv2, (short)0, acc1, false, false);
        acc1 = __builtin_amdgcn_wmma_f32_16x16x32_f16(false, a1, false, bv3, (short)0, acc1, false, false);
#pragma unroll
        for (int nt = 0; nt < 2; ++nt) {
            const v8f acc = nt ? acc1 : acc0;
            const int dc = nt * 16 + (lane & 15);
            _Float16* op = &Ost[(mt * 16 + 8 * hi) * CDIM + h * DHEAD + dc];
#pragma unroll
            for (int j = 0; j < 8; ++j)
                op[j * CDIM] = (_Float16)acc[j];
        }
    }
    __syncthreads();

    // ---- Phase 3: out = O @ proj_w^T + proj_b (f32 store, rows < 49) ----
    // Each wave owns one N-tile (nt == wave, scalar); B fragments hoisted, reused over 4 M-tiles.
    {
        const int nt = wave;
        v16h bf0 = lds_load_BT(wP, CDIM, nt * 16, 0,  lane);
        v16h bf1 = lds_load_BT(wP, CDIM, nt * 16, 32, lane);
        v16h bf2 = lds_load_BT(wP, CDIM, nt * 16, 64, lane);
        v16h bf3 = lds_load_BT(wP, CDIM, nt * 16, 96, lane);
        const int n  = nt * 16 + (lane & 15);
        const int hi = (lane >> 4) & 1;
        const float bn = bP[n];
        float* ob = out + (size_t)b * LWIN * CDIM;
#pragma unroll 1
        for (int mt = 0; mt < 4; ++mt) {
            const v16h a0 = lds_load_A(Ost, CDIM, mt * 16, 0,  lane);
            const v16h a1 = lds_load_A(Ost, CDIM, mt * 16, 32, lane);
            const v16h a2 = lds_load_A(Ost, CDIM, mt * 16, 64, lane);
            const v16h a3 = lds_load_A(Ost, CDIM, mt * 16, 96, lane);
            v8f acc = {};
            acc = __builtin_amdgcn_wmma_f32_16x16x32_f16(false, a0, false, bf0, (short)0, acc, false, false);
            acc = __builtin_amdgcn_wmma_f32_16x16x32_f16(false, a1, false, bf1, (short)0, acc, false, false);
            acc = __builtin_amdgcn_wmma_f32_16x16x32_f16(false, a2, false, bf2, (short)0, acc, false, false);
            acc = __builtin_amdgcn_wmma_f32_16x16x32_f16(false, a3, false, bf3, (short)0, acc, false, false);
#pragma unroll
            for (int j = 0; j < 8; ++j) {
                const int m = mt * 16 + j + 8 * hi;
                if (m < LWIN) ob[m * CDIM + n] = acc[j] + bn;
            }
        }
    }
}

extern "C" void kernel_launch(void* const* d_in, const int* in_sizes, int n_in,
                              void* d_out, int out_size, void* d_ws, size_t ws_size,
                              hipStream_t stream) {
    const float* x      = (const float*)d_in[0];
    const float* mask   = (const float*)d_in[1];
    const float* qkv_w  = (const float*)d_in[2];
    const float* qkv_b  = (const float*)d_in[3];
    const float* bias   = (const float*)d_in[4];
    const float* proj_w = (const float*)d_in[5];
    const float* proj_b = (const float*)d_in[6];
    float* out = (float*)d_out;

    const int B = 8192;                 // in_sizes[0] = B*L*C = 8192*49*128
    swin_mha_fused<<<B, 256, 0, stream>>>(x, mask, qkv_w, qkv_b, bias,
                                          proj_w, proj_b, out);
}
